// Multi_Adaptation_Module_18717467476190
// MI455X (gfx1250) — matrix operand order repaired and checked by reading, not compiled, not run on hardware
//
#include <hip/hip_runtime.h>
#include <hip/hip_bf16.h>

#define CCH  512
#define NPIX 4096
#define BATCH 4

typedef __attribute__((ext_vector_type(16))) __bf16 v16bf;
typedef __attribute__((ext_vector_type(8)))  float  v8f;

union Frag { v16bf v; unsigned int u[8]; };

__device__ __forceinline__ unsigned short f2bf(float f) {
    unsigned int x = __float_as_uint(f);
    unsigned int r = (x + 0x7FFFu + ((x >> 16) & 1u)) >> 16;
    return (unsigned short)r;
}

// ---------------------------------------------------------------------------
// f32 -> bf16 flat convert (weights)
// ---------------------------------------------------------------------------
__global__ void cvt_f32_bf16(const float* __restrict__ src,
                             unsigned short* __restrict__ dst, int n) {
    int i = blockIdx.x * blockDim.x + threadIdx.x;
    int stride = gridDim.x * blockDim.x;
    for (; i < n; i += stride) dst[i] = f2bf(src[i]);
}

// ---------------------------------------------------------------------------
// Per-(b,c) mean-variance norm (unbiased, eps=1e-5), writing bf16 outputs
// transposed to [B, N, C] (fragment-friendly layout). Xt may be null.
// One block per (b,c), 256 threads, N=4096.
// ---------------------------------------------------------------------------
__global__ void mvn_transpose(const float* __restrict__ X,
                              unsigned short* __restrict__ Xnt,
                              unsigned short* __restrict__ Xt) {
    __shared__ float s1[256], s2[256];
    int b = blockIdx.x / CCH, c = blockIdx.x % CCH;
    const float* xp = X + ((size_t)b * CCH + c) * NPIX;
    float v[16];
    float sum = 0.f, sq = 0.f;
#pragma unroll
    for (int j = 0; j < 16; ++j) {
        float t = xp[threadIdx.x + j * 256];
        v[j] = t; sum += t; sq += t * t;
    }
    s1[threadIdx.x] = sum; s2[threadIdx.x] = sq;
    __syncthreads();
    for (int off = 128; off > 0; off >>= 1) {
        if ((int)threadIdx.x < off) {
            s1[threadIdx.x] += s1[threadIdx.x + off];
            s2[threadIdx.x] += s2[threadIdx.x + off];
        }
        __syncthreads();
    }
    float mean = s1[0] * (1.0f / NPIX);
    float var  = (s2[0] - s1[0] * mean) * (1.0f / (NPIX - 1));
    float rstd = rsqrtf(var + 1e-5f);
    unsigned short* on = Xnt + (size_t)b * NPIX * CCH + c;
#pragma unroll
    for (int j = 0; j < 16; ++j) {
        int p = threadIdx.x + j * 256;
        on[(size_t)p * CCH] = f2bf((v[j] - mean) * rstd);
    }
    if (Xt) {
        unsigned short* ot = Xt + (size_t)b * NPIX * CCH + c;
#pragma unroll
        for (int j = 0; j < 16; ++j) {
            int p = threadIdx.x + j * 256;
            ot[(size_t)p * CCH] = f2bf(v[j]);
        }
    }
}

// ---------------------------------------------------------------------------
// Generic WMMA GEMM:  Y[b] = A[b] (512 x K, bf16 row-major) * B[b]
// where Bt is stored "transposed" row-major [Nout, K] bf16 so every B-fragment
// pair load is contiguous.  outMode: 0 = bf16 transposed [Nout,512],
// 1 = bf16 [512,Nout], 2 = f32 [512,Nout] (+ optional f32 residual).
// Block: 256 threads (8 waves), tile 128(M) x 128(N), K step 32.
// ---------------------------------------------------------------------------
__global__ void gemm512(const unsigned short* __restrict__ A, size_t aBatch,
                        const unsigned short* __restrict__ Bt, size_t bBatch,
                        int K, int Nout,
                        const float* __restrict__ bias,
                        int outMode,
                        void* __restrict__ Yv, size_t yBatch,
                        const float* __restrict__ Res, size_t rBatch) {
    int b = blockIdx.z;
    const unsigned short* Ab = A + (size_t)b * aBatch;
    const unsigned short* Bb = Bt + (size_t)b * bBatch;
    int w  = threadIdx.x >> 5;
    int l  = threadIdx.x & 31;
    int lm = l & 15, lh = l >> 4;
    int wm = w >> 2, wn = w & 3;
    int mBase = blockIdx.y * 128 + wm * 64;   // 2 waves x 64 rows
    int nBase = blockIdx.x * 128 + wn * 32;   // 4 waves x 32 cols

    v8f acc[4][2];
#pragma unroll
    for (int mt = 0; mt < 4; ++mt)
#pragma unroll
        for (int nt = 0; nt < 2; ++nt)
#pragma unroll
            for (int r = 0; r < 8; ++r) acc[mt][nt][r] = 0.f;

    for (int kk = 0; kk < K; kk += 32) {
        Frag af[4], bf[2];
#pragma unroll
        for (int mt = 0; mt < 4; ++mt) {
            const unsigned short* ap = Ab + (size_t)(mBase + mt * 16 + lm) * K + kk;
#pragma unroll
            for (int i = 0; i < 8; ++i) {
                int ko = lh * 16 + 2 * i;
                af[mt].u[i] = *(const unsigned int*)(ap + ko);
            }
        }
#pragma unroll
        for (int nt = 0; nt < 2; ++nt) {
            const unsigned short* bp = Bb + (size_t)(nBase + nt * 16 + lm) * K + kk + lh * 16;
#pragma unroll
            for (int i = 0; i < 8; ++i)
                bf[nt].u[i] = *(const unsigned int*)(bp + 2 * i);
        }
#pragma unroll
        for (int mt = 0; mt < 4; ++mt)
#pragma unroll
            for (int nt = 0; nt < 2; ++nt)
                acc[mt][nt] = __builtin_amdgcn_wmma_f32_16x16x32_bf16(
                    false, af[mt].v, false, bf[nt].v, (short)0, acc[mt][nt], false, false);
    }

#pragma unroll
    for (int mt = 0; mt < 4; ++mt) {
#pragma unroll
        for (int nt = 0; nt < 2; ++nt) {
            int n = nBase + nt * 16 + lm;
#pragma unroll
            for (int r = 0; r < 8; ++r) {
                int m = mBase + mt * 16 + r + lh * 8;
                float val = acc[mt][nt][r];
                if (bias) val += bias[m];
                if (outMode == 0) {
                    ((unsigned short*)Yv)[(size_t)b * yBatch + (size_t)n * CCH + m] = f2bf(val);
                } else if (outMode == 1) {
                    ((unsigned short*)Yv)[(size_t)b * yBatch + (size_t)m * Nout + n] = f2bf(val);
                } else {
                    if (Res) val += Res[(size_t)b * rBatch + (size_t)m * Nout + n];
                    ((float*)Yv)[(size_t)b * yBatch + (size_t)m * Nout + n] = val;
                }
            }
        }
    }
}

// ---------------------------------------------------------------------------
// Flash attention over pixels:  O[i,c] = softmax_j( q_i . k_j ) * v[c,j]
// Qt, Kt : [B, N, 512] bf16 (transposed), Vnt : [B, 512, N] bf16,
// Ot : [B, N, 512] bf16.  Block = 16 queries x all 512 channels, 8 waves
// each owning 64 channels; streams 128 keys per iteration with online
// softmax staged through LDS.
// ---------------------------------------------------------------------------
#define SSP 132
__global__ void flash_attn(const unsigned short* __restrict__ Qt,
                           const unsigned short* __restrict__ Kt,
                           const unsigned short* __restrict__ Vnt,
                           unsigned short* __restrict__ Ot,
                           int Nq, int Nk) {
    __shared__ float sS[16][SSP];
    __shared__ unsigned short sP[16][128];
    __shared__ float sMax[16], sSum[16], sScale[16], sRed[16][16];

    int b = blockIdx.y;
    int qBase = blockIdx.x * 16;
    int w = threadIdx.x >> 5, l = threadIdx.x & 31;
    int lm = l & 15, lh = l >> 4;
    const unsigned short* Qb = Qt + (size_t)b * Nq * CCH;
    const unsigned short* Kb = Kt + (size_t)b * Nk * CCH;
    const unsigned short* Vb = Vnt + (size_t)b * CCH * Nk;

    v8f acc[4];
#pragma unroll
    for (int t = 0; t < 4; ++t)
#pragma unroll
        for (int r = 0; r < 8; ++r) acc[t][r] = 0.f;
    if (threadIdx.x < 16) { sMax[threadIdx.x] = -3.0e38f; sSum[threadIdx.x] = 0.f; }
    __syncthreads();

    int row2 = threadIdx.x & 15;
    int seg  = threadIdx.x >> 4;

    const unsigned short* qp0 = Qb + (size_t)(qBase + lm) * CCH;

    for (int jb = 0; jb < Nk; jb += 128) {
        // ---- S tile: this wave handles keys [jb + w*16, +16) over full C ----
        v8f s;
#pragma unroll
        for (int r = 0; r < 8; ++r) s[r] = 0.f;
        const unsigned short* kp0 = Kb + (size_t)(jb + w * 16 + lm) * CCH;
        for (int kk = 0; kk < CCH; kk += 32) {
            Frag a, bf;
#pragma unroll
            for (int i = 0; i < 8; ++i) {
                int ko = kk + lh * 16 + 2 * i;
                a.u[i] = *(const unsigned int*)(qp0 + ko);
            }
#pragma unroll
            for (int i = 0; i < 8; ++i)
                bf.u[i] = *(const unsigned int*)(kp0 + kk + lh * 16 + 2 * i);
            s = __builtin_amdgcn_wmma_f32_16x16x32_bf16(
                false, a.v, false, bf.v, (short)0, s, false, false);
        }
#pragma unroll
        for (int r = 0; r < 8; ++r)
            sS[r + lh * 8][w * 16 + lm] = s[r];
        __syncthreads();

        // ---- online softmax over the 128 keys of this iteration ----
        float pm = -3.0e38f;
#pragma unroll
        for (int j = 0; j < 8; ++j) pm = fmaxf(pm, sS[row2][seg * 8 + j]);
        sRed[row2][seg] = pm;
        __syncthreads();
        if (threadIdx.x < 16) {
            float tm = sRed[threadIdx.x][0];
#pragma unroll
            for (int j = 1; j < 16; ++j) tm = fmaxf(tm, sRed[threadIdx.x][j]);
            float nm = fmaxf(sMax[threadIdx.x], tm);
            sScale[threadIdx.x] = __expf(sMax[threadIdx.x] - nm);
            sMax[threadIdx.x] = nm;
        }
        __syncthreads();
        float rm = sMax[row2];
        float ps = 0.f;
#pragma unroll
        for (int j = 0; j < 8; ++j) {
            float e = __expf(sS[row2][seg * 8 + j] - rm);
            sP[row2][seg * 8 + j] = f2bf(e);
            ps += e;
        }
        sRed[row2][seg] = ps;
        __syncthreads();
        if (threadIdx.x < 16) {
            float ts = 0.f;
#pragma unroll
            for (int j = 0; j < 16; ++j) ts += sRed[threadIdx.x][j];
            sSum[threadIdx.x] = sSum[threadIdx.x] * sScale[threadIdx.x] + ts;
        }
        __syncthreads();

        // ---- rescale running O, then O += P * V^T for this key block ----
#pragma unroll
        for (int r = 0; r < 8; ++r) {
            float f = sScale[r + lh * 8];
#pragma unroll
            for (int t = 0; t < 4; ++t) acc[t][r] *= f;
        }
        for (int ks = 0; ks < 128; ks += 32) {
            Frag a;
#pragma unroll
            for (int i = 0; i < 8; ++i) {
                int ko = ks + lh * 16 + 2 * i;
                a.u[i] = *(const unsigned int*)(&sP[lm][ko]);
            }
#pragma unroll
            for (int t = 0; t < 4; ++t) {
                Frag bv;
                int c = w * 64 + t * 16 + lm;
                const unsigned short* vp = Vb + (size_t)c * Nk + jb + ks + lh * 16;
#pragma unroll
                for (int i = 0; i < 8; ++i)
                    bv.u[i] = *(const unsigned int*)(vp + 2 * i);
                acc[t] = __builtin_amdgcn_wmma_f32_16x16x32_bf16(
                    false, a.v, false, bv.v, (short)0, acc[t], false, false);
            }
        }
        __syncthreads();
    }

#pragma unroll
    for (int r = 0; r < 8; ++r) {
        int qi = qBase + r + lh * 8;
        float inv = 1.0f / sSum[r + lh * 8];
#pragma unroll
        for (int t = 0; t < 4; ++t) {
            int c = w * 64 + t * 16 + lm;
            Ot[(size_t)b * Nq * CCH + (size_t)qi * CCH + c] = f2bf(acc[t][r] * inv);
        }
    }
}

// ---------------------------------------------------------------------------
// Row softmax of E [B, C, C] f32 (over last dim), written bf16 TRANSPOSED:
// AttnT[b, col, row] = softmax(E[b,row,:])[col].  One block per (b,row).
// ---------------------------------------------------------------------------
__global__ void softmax_rows_t(const float* __restrict__ E,
                               unsigned short* __restrict__ AttnT) {
    __shared__ float sred[256];
    int b = blockIdx.x / CCH, row = blockIdx.x % CCH;
    const float* ep = E + ((size_t)b * CCH + row) * CCH;
    float v0 = ep[threadIdx.x], v1 = ep[threadIdx.x + 256];
    sred[threadIdx.x] = fmaxf(v0, v1);
    __syncthreads();
    for (int off = 128; off > 0; off >>= 1) {
        if ((int)threadIdx.x < off)
            sred[threadIdx.x] = fmaxf(sred[threadIdx.x], sred[threadIdx.x + off]);
        __syncthreads();
    }
    float rm = sred[0];
    __syncthreads();
    float e0 = __expf(v0 - rm), e1 = __expf(v1 - rm);
    sred[threadIdx.x] = e0 + e1;
    __syncthreads();
    for (int off = 128; off > 0; off >>= 1) {
        if ((int)threadIdx.x < off) sred[threadIdx.x] += sred[threadIdx.x + off];
        __syncthreads();
    }
    float inv = 1.0f / sred[0];
    unsigned short* op = AttnT + (size_t)b * CCH * CCH + row;
    op[(size_t)threadIdx.x * CCH]         = f2bf(e0 * inv);
    op[(size_t)(threadIdx.x + 256) * CCH] = f2bf(e1 * inv);
}

// ---------------------------------------------------------------------------
extern "C" void kernel_launch(void* const* d_in, const int* in_sizes, int n_in,
                              void* d_out, int out_size, void* d_ws, size_t ws_size,
                              hipStream_t stream) {
    (void)in_sizes; (void)n_in; (void)out_size; (void)ws_size;
    const float* content = (const float*)d_in[0];
    const float* style   = (const float*)d_in[1];
    const float* csa_w = (const float*)d_in[2];
    const float* csa_b = (const float*)d_in[3];
    const float* ssa_w = (const float*)d_in[4];
    const float* ssa_b = (const float*)d_in[5];
    const float* ca_w  = (const float*)d_in[6];
    const float* ca_b  = (const float*)d_in[7];
    float* out = (float*)d_out;

    // workspace layout
    char* ws = (char*)d_ws;
    const size_t bufE = (size_t)NPIX * CCH;               // per-batch elems
    const size_t bufBbytes = (size_t)BATCH * bufE * 2;    // bf16 buffer bytes
    unsigned short* B[7];
    for (int i = 0; i < 7; ++i) B[i] = (unsigned short*)(ws + (size_t)i * bufBbytes);
    float* F0 = (float*)(ws + 7 * bufBbytes);                              // cf f32 [B,C,N]
    float* F1 = (float*)((char*)F0 + (size_t)BATCH * CCH * NPIX * 4);      // sf f32 [B,C,N]
    float* Ebuf = (float*)((char*)F1 + (size_t)BATCH * CCH * NPIX * 4);    // [B,C,C] f32
    unsigned short* Wb = (unsigned short*)((char*)Ebuf + (size_t)BATCH * CCH * CCH * 4);

    const size_t WC = (size_t)CCH * CCH;
    const size_t CN = (size_t)CCH * NPIX;
    auto Wp = [&](int idx) { return Wb + (size_t)idx * WC; };

    dim3 blk(256);
    dim3 gConv(NPIX / 128, CCH / 128, BATCH);
    dim3 gE(CCH / 128, CCH / 128, BATCH);
    dim3 gAttn(NPIX / 16, BATCH);
    int nW = 4 * CCH * CCH;

    // weights -> bf16 (csa:0-3, ssa:4-7, ca:8-11)
    cvt_f32_bf16<<<256, blk, 0, stream>>>(csa_w, Wp(0), nW);
    cvt_f32_bf16<<<256, blk, 0, stream>>>(ssa_w, Wp(4), nW);
    cvt_f32_bf16<<<256, blk, 0, stream>>>(ca_w,  Wp(8), nW);

    // ---------------- content self-attention ----------------
    mvn_transpose<<<BATCH * CCH, blk, 0, stream>>>(content, B[0], B[1]); // xnt, xt
    gemm512<<<gConv, blk, 0, stream>>>(Wp(0), 0, B[0], bufE, CCH, NPIX, csa_b + 0 * CCH, 0, B[2], bufE, nullptr, 0); // q (T)
    gemm512<<<gConv, blk, 0, stream>>>(Wp(1), 0, B[0], bufE, CCH, NPIX, csa_b + 1 * CCH, 0, B[3], bufE, nullptr, 0); // k (T)
    gemm512<<<gConv, blk, 0, stream>>>(Wp(2), 0, B[1], bufE, CCH, NPIX, csa_b + 2 * CCH, 1, B[4], bufE, nullptr, 0); // v (NT)
    flash_attn<<<gAttn, blk, 0, stream>>>(B[2], B[3], B[4], B[5], NPIX, NPIX);                                       // o (T)
    gemm512<<<gConv, blk, 0, stream>>>(Wp(3), 0, B[5], bufE, CCH, NPIX, csa_b + 3 * CCH, 2, F0, CN, content, CN);    // cf f32

    // ---------------- style self-attention (channel attn) ----------------
    mvn_transpose<<<BATCH * CCH, blk, 0, stream>>>(style, B[0], B[1]);   // xnt, xt
    gemm512<<<gConv, blk, 0, stream>>>(Wp(4), 0, B[1], bufE, CCH, NPIX, ssa_b + 0 * CCH, 1, B[2], bufE, nullptr, 0); // f raw (NT)
    gemm512<<<gConv, blk, 0, stream>>>(Wp(5), 0, B[1], bufE, CCH, NPIX, ssa_b + 1 * CCH, 1, B[3], bufE, nullptr, 0); // g raw (NT)
    gemm512<<<gE,    blk, 0, stream>>>(B[2], bufE, B[3], bufE, NPIX, CCH, nullptr, 2, Ebuf, WC, nullptr, 0);         // energy f32
    softmax_rows_t<<<BATCH * CCH, blk, 0, stream>>>(Ebuf, B[4]);                                                     // attnT bf16
    gemm512<<<gConv, blk, 0, stream>>>(Wp(6), 0, B[0], bufE, CCH, NPIX, ssa_b + 2 * CCH, 0, B[5], bufE, nullptr, 0); // h = conv(mvn) (T)
    gemm512<<<gConv, blk, 0, stream>>>(B[4], WC, B[5], bufE, CCH, NPIX, nullptr, 0, B[2], bufE, nullptr, 0);         // attn^T @ h (T)
    gemm512<<<gConv, blk, 0, stream>>>(Wp(7), 0, B[2], bufE, CCH, NPIX, ssa_b + 3 * CCH, 2, F1, CN, style, CN);      // sf f32

    // ---------------- cross attention ----------------
    mvn_transpose<<<BATCH * CCH, blk, 0, stream>>>(F0, B[0], nullptr);   // mvn(cf)
    mvn_transpose<<<BATCH * CCH, blk, 0, stream>>>(F1, B[1], B[3]);      // mvn(sf), sf raw
    gemm512<<<gConv, blk, 0, stream>>>(Wp(8),  0, B[0], bufE, CCH, NPIX, ca_b + 0 * CCH, 0, B[2], bufE, nullptr, 0); // q (T)
    gemm512<<<gConv, blk, 0, stream>>>(Wp(9),  0, B[1], bufE, CCH, NPIX, ca_b + 1 * CCH, 0, B[4], bufE, nullptr, 0); // k (T)
    gemm512<<<gConv, blk, 0, stream>>>(Wp(10), 0, B[3], bufE, CCH, NPIX, ca_b + 2 * CCH, 1, B[5], bufE, nullptr, 0); // v (NT)
    flash_attn<<<gAttn, blk, 0, stream>>>(B[2], B[4], B[5], B[6], NPIX, NPIX);                                       // o (T)
    gemm512<<<gConv, blk, 0, stream>>>(Wp(11), 0, B[6], bufE, CCH, NPIX, ca_b + 3 * CCH, 2, out, CN, F0, CN);        // final
}
